// HACContextModel_69372311765527
// MI455X (gfx1250) — matrix operand order repaired and checked by reading, not compiled, not run on hardware
//
#include <hip/hip_runtime.h>
#include <hip/hip_bf16.h>
#include <math.h>

typedef __attribute__((ext_vector_type(16))) _Float16 v16h;
typedef __attribute__((ext_vector_type(8)))  _Float16 v8h;
typedef __attribute__((ext_vector_type(8)))  float    v8f;

// ---------------- problem constants ----------------
// N points, H=48 hash feats, A=32 anchor, S=6 scaling, OD=30 offset, TOT=68
// CS=8 per context chunk, K=3 mixture comps (6 for chunks 1..3)

// ---------------- f16 weight arena layout (offsets in halves) ----------------
// storage: transposed [Nout_pad][Kpad], zero padded
static constexpr size_t OFF_WQ1 = 0;                       // 128 x 64
static constexpr size_t OFF_WQ2 = OFF_WQ1 + 128 * 64;      // 80 x 128
static constexpr size_t OFF_WC1 = OFF_WQ2 + 80 * 128;      // 4 x (128 x 64)
static constexpr size_t OFF_WC2 = OFF_WC1 + 4 * 128 * 64;  // 4 x (80 x 128)
static constexpr size_t OFF_WS1 = OFF_WC2 + 4 * 80 * 128;  // 64 x 64
static constexpr size_t OFF_WS2 = OFF_WS1 + 64 * 64;       // 64 x 64
static constexpr size_t OFF_WO1 = OFF_WS2 + 64 * 64;       // 128 x 64
static constexpr size_t OFF_WO2 = OFF_WO1 + 128 * 64;      // 272 x 128
static constexpr size_t OFF_WA1 = OFF_WO2 + 272 * 128;     // 3 x (128 x 96)
static constexpr size_t OFF_WA2 = OFF_WA1 + 3 * 128 * 96;  // 3 x (80 x 128)
static constexpr size_t WT_TOTAL = OFF_WA2 + 3 * 80 * 128; // 210944 halves

// ---------------- weight prep: fp32 [K][N] -> f16 transposed [Npad][Kpad] ----
struct PrepEntry { const float* src; int K; int N; int Kpad; size_t off; size_t cnt; };
struct PrepTable { PrepEntry e[20]; };

__global__ void hac_prep_weights(PrepTable t, _Float16* wt, int total) {
  int gid = blockIdx.x * blockDim.x + threadIdx.x;
  if (gid >= total) return;
  size_t g = (size_t)gid;
  for (int i = 0; i < 20; ++i) {
    const PrepEntry en = t.e[i];
    if (g >= en.off && g < en.off + en.cnt) {
      size_t local = g - en.off;
      int nrow = (int)(local / (size_t)en.Kpad);
      int k = (int)(local % (size_t)en.Kpad);
      float v = 0.f;
      if (k < en.K && nrow < en.N) v = en.src[(size_t)k * en.N + nrow];
      wt[g] = (_Float16)v;
      return;
    }
  }
}

// ---------------- WMMA fragment loaders (16x16x32 f16; this code's contiguous k order) ------
// A (16x32 f16): lane m=l&15; K set = (l<16)? {0..15} : {16..31}
__device__ __forceinline__ v16h load_A_frag(const _Float16* buf, int stride, int kt, int lane) {
  int m = lane & 15, hi = lane >> 4;
  const _Float16* r = buf + m * stride + kt * 32;
  v8h lo = *(const v8h*)(r + hi * 16);
  v8h hh = *(const v8h*)(r + hi * 16 + 8);
  return __builtin_shufflevector(lo, hh, 0,1,2,3,4,5,6,7,8,9,10,11,12,13,14,15);
}
// B (32x16 f16): lanes 0-15 hold col n=lane, K=0..15; lanes 16-31 K=16..31.
// weights stored transposed [n][Kpad] -> one contiguous 32B read per lane.
__device__ __forceinline__ v16h load_B_frag(const _Float16* wt, int Kpad, int nt, int kt, int lane) {
  int nn = lane & 15, hi = lane >> 4;
  return *(const v16h*)(wt + (size_t)(nt * 16 + nn) * Kpad + (size_t)kt * 32 + hi * 16);
}

// One GEMM layer for a 16-point tile:  D[16][16*ntiles] = A[16][32*ksteps] x W
// out_h != nullptr : relu -> f16 (layer 1).  else: +bias f32 -> out_f (layer 2)
__device__ __forceinline__ void run_layer(
    const _Float16* inbuf, int inStride, int ksteps,
    const _Float16* wt, int Kpad, const float* bias, int nReal,
    int nt0, int ntiles, int lane,
    _Float16* out_h, int outStrideH, float* out_f, int outStrideF) {
  v16h afr[4];
  #pragma unroll
  for (int kt = 0; kt < 4; ++kt)
    if (kt < ksteps) afr[kt] = load_A_frag(inbuf, inStride, kt, lane);
  int nn = lane & 15, hi = lane >> 4;
  for (int t = 0; t < ntiles; ++t) {
    int nt = nt0 + t;
    v8f acc = {0.f,0.f,0.f,0.f,0.f,0.f,0.f,0.f};
    for (int kt = 0; kt < ksteps; ++kt) {
      v16h b = load_B_frag(wt, Kpad, nt, kt, lane);
      acc = __builtin_amdgcn_wmma_f32_16x16x32_f16(
          /*neg_a=*/false, afr[kt], /*neg_b=*/false, b,
          /*c_mod=*/(short)0, acc, /*reuse_a=*/false, /*reuse_b=*/false);
    }
    int col = nt * 16 + nn;
    float bv = (col < nReal) ? bias[col] : 0.f;
    // D layout: lane holds col=nn for rows M = hi*8 .. hi*8+7
    if (out_h) {
      #pragma unroll
      for (int r = 0; r < 8; ++r)
        out_h[(hi * 8 + r) * outStrideH + col] = (_Float16)fmaxf(acc[r] + bv, 0.f);
    } else {
      int lcol = t * 16 + nn;
      #pragma unroll
      for (int r = 0; r < 8; ++r)
        out_f[(hi * 8 + r) * outStrideF + lcol] = acc[r] + bv;
    }
  }
}

// ---------------- GMM bits ----------------
__device__ __forceinline__ float softplusf(float x) {
  return (x > 20.f) ? x : log1pf(expf(x));
}
__device__ __forceinline__ float gmm_bits(float x, float q, const float* mu,
                                          const float* sg, const float* pw, int K) {
  float mx = pw[0];
  for (int k = 1; k < K; ++k) mx = fmaxf(mx, pw[k]);
  float w[6], wsum = 0.f;
  for (int k = 0; k < K; ++k) { w[k] = expf(pw[k] - mx); wsum += w[k]; }
  float inv = 1.f / wsum;
  const float is2 = 0.7071067811865476f;
  float prob = 0.f;
  for (int k = 0; k < K; ++k) {
    float sig = softplusf(sg[k]) + 1e-6f;
    float rs = is2 / sig;
    float up = erff((x + 0.5f * q - mu[k]) * rs);
    float lo = erff((x - 0.5f * q - mu[k]) * rs);
    float pk = fmaxf(0.5f * (up - lo), 1e-10f);
    prob += (w[k] * inv) * pk;
  }
  prob = fmaxf(prob, 1e-10f);
  return -log2f(prob);
}

// ---------------- main kernel ----------------
struct HacArgs {
  const float *hash, *anchor, *scaling, *offset, *u_feat, *u_scaling, *u_offset;
  const float *bq1, *bq2, *bc1, *bc2, *bs1, *bs2, *bo1, *bo2;
  const float *ba1[3], *ba2[3];
};

#define WAVE_BYTES 27136  // per-wave LDS partition

__global__ void __launch_bounds__(64)
hac_main(HacArgs Ag, const _Float16* __restrict__ wt, float* __restrict__ out, int n) {
  extern __shared__ char smem[];
  const int lane = threadIdx.x & 31;
  const int wave = threadIdx.x >> 5;
  char* wb = smem + wave * WAVE_BYTES;
  _Float16* conc = (_Float16*)wb;              // [16][96]  hash(48)|feat_hat(32)|0(16)
  _Float16* hbuf = (_Float16*)(wb + 3072);     // [16][128] hidden acts (f16)
  float*    qv   = (float*)(wb + 7168);        // [16][80]  q vector
  float*    hat  = (float*)(wb + 12288);       // [16][72]  feat|scal|off hats
  float*    scr  = (float*)(wb + 16896);       // [16][160] layer-2 f32 scratch

  const int pbase = blockIdx.x * 32 + wave * 16;

  // ---- load hash feats, zero pad region ----
  for (int e = lane; e < 16 * 48; e += 32) {
    int p = e / 48, c = e - p * 48;
    int idx = pbase + p; if (idx >= n) idx = n - 1;
    conc[p * 96 + c]      = (_Float16)Ag.hash[(size_t)idx * 48 + c];
    conc[p * 96 + 48 + c] = (_Float16)0.f;
  }
  __syncthreads();

  // ---- q MLP: 48->128 relu ->68, then q = s*(1+tanh(r)) ----
  run_layer(conc, 96, 2, wt + OFF_WQ1, 64, Ag.bq1, 128, 0, 8, lane, hbuf, 128, nullptr, 0);
  __syncthreads();
  run_layer(hbuf, 128, 4, wt + OFF_WQ2, 128, Ag.bq2, 68, 0, 5, lane, nullptr, 0, qv, 80);
  __syncthreads();
  for (int e = lane; e < 16 * 68; e += 32) {
    int p = e / 68, c = e - p * 68;
    float s = (c < 32) ? 1.0f : (c < 38 ? 0.001f : 0.2f);
    qv[p * 80 + c] = s * (1.f + tanhf(qv[p * 80 + c]));
  }
  __syncthreads();

  // ---- hats (also stream to output; feat_hat also as f16 into conc) ----
  for (int e = lane; e < 16 * 32; e += 32) {
    int p = e >> 5, c = e & 31;
    int gp = pbase + p; bool ok = gp < n; int idx = ok ? gp : n - 1;
    float fh = Ag.anchor[(size_t)idx * 32 + c] +
               (Ag.u_feat[(size_t)idx * 32 + c] - 0.5f) * qv[p * 80 + c];
    hat[p * 72 + c] = fh;
    conc[p * 96 + 48 + c] = (_Float16)fh;
    if (ok) out[(size_t)3 * n + (size_t)gp * 32 + c] = fh;
  }
  for (int e = lane; e < 16 * 6; e += 32) {
    int p = e / 6, c = e - p * 6;
    int gp = pbase + p; bool ok = gp < n; int idx = ok ? gp : n - 1;
    float sh = Ag.scaling[(size_t)idx * 6 + c] +
               (Ag.u_scaling[(size_t)idx * 6 + c] - 0.5f) * qv[p * 80 + 32 + c];
    hat[p * 72 + 32 + c] = sh;
    if (ok) out[(size_t)35 * n + (size_t)gp * 6 + c] = sh;
  }
  for (int e = lane; e < 16 * 30; e += 32) {
    int p = e / 30, c = e - p * 30;
    int gp = pbase + p; bool ok = gp < n; int idx = ok ? gp : n - 1;
    float oh = Ag.offset[(size_t)idx * 30 + c] +
               (Ag.u_offset[(size_t)idx * 30 + c] - 0.5f) * qv[p * 80 + 38 + c];
    hat[p * 72 + 38 + c] = oh;
    if (ok) out[(size_t)41 * n + (size_t)gp * 30 + c] = oh;
  }
  __syncthreads();

  const int p = lane & 15, hf = lane >> 4;
  float bitsF = 0.f, bitsS = 0.f, bitsO = 0.f;

  // ---- feature chunks: ctx MLP (+ intra MLP for i>0) then GMM ----
  for (int ci = 0; ci < 4; ++ci) {
    run_layer(conc, 96, 2, wt + OFF_WC1 + (size_t)ci * 128 * 64, 64,
              Ag.bc1 + ci * 128, 128, 0, 8, lane, hbuf, 128, nullptr, 0);
    __syncthreads();
    run_layer(hbuf, 128, 4, wt + OFF_WC2 + (size_t)ci * 80 * 128, 128,
              Ag.bc2 + ci * 72, 72, 0, 5, lane, nullptr, 0, scr, 160);
    __syncthreads();
    if (ci > 0) {
      run_layer(conc, 96, 3, wt + OFF_WA1 + (size_t)(ci - 1) * 128 * 96, 96,
                Ag.ba1[ci - 1], 128, 0, 8, lane, hbuf, 128, nullptr, 0);
      __syncthreads();
      run_layer(hbuf, 128, 4, wt + OFF_WA2 + (size_t)(ci - 1) * 80 * 128, 128,
                Ag.ba2[ci - 1], 72, 0, 5, lane, nullptr, 0, scr + 80, 160);
      __syncthreads();
    }
    int Kc = ci ? 6 : 3;
    for (int dd = 0; dd < 4; ++dd) {
      int d = hf * 4 + dd;                        // chunk-local dim 0..7
      float x = hat[p * 72 + ci * 8 + d];
      float q = qv[p * 80 + ci * 8 + d];
      float mu[6], sg[6], pw[6];
      for (int k = 0; k < 3; ++k) {
        int b = p * 160 + d * 9 + k * 3;
        mu[k] = scr[b]; sg[k] = scr[b + 1]; pw[k] = scr[b + 2];
      }
      if (ci) {
        for (int k = 0; k < 3; ++k) {
          int b = p * 160 + 80 + d * 9 + k * 3;
          mu[3 + k] = scr[b]; sg[3 + k] = scr[b + 1]; pw[3 + k] = scr[b + 2];
        }
      }
      bitsF += gmm_bits(x, q, mu, sg, pw, Kc);
    }
    __syncthreads();
  }

  // ---- scaling head: 48->64 relu ->54 ----
  run_layer(conc, 96, 2, wt + OFF_WS1, 64, Ag.bs1, 64, 0, 4, lane, hbuf, 128, nullptr, 0);
  __syncthreads();
  run_layer(hbuf, 128, 2, wt + OFF_WS2, 64, Ag.bs2, 54, 0, 4, lane, nullptr, 0, scr, 160);
  __syncthreads();
  for (int dd = 0; dd < 3; ++dd) {
    int d = hf * 3 + dd;                          // 0..5
    float x = hat[p * 72 + 32 + d], q = qv[p * 80 + 32 + d];
    float mu[6], sg[6], pw[6];
    for (int k = 0; k < 3; ++k) {
      int b = p * 160 + d * 9 + k * 3;
      mu[k] = scr[b]; sg[k] = scr[b + 1]; pw[k] = scr[b + 2];
    }
    bitsS += gmm_bits(x, q, mu, sg, pw, 3);
  }
  __syncthreads();

  // ---- offset head: 48->128 relu ->270, processed in two column halves ----
  run_layer(conc, 96, 2, wt + OFF_WO1, 64, Ag.bo1, 128, 0, 8, lane, hbuf, 128, nullptr, 0);
  __syncthreads();
  run_layer(hbuf, 128, 4, wt + OFF_WO2, 128, Ag.bo2, 270, 0, 9, lane, nullptr, 0, scr, 160);
  __syncthreads();
  for (int dd = 0; dd < 8; ++dd) {
    int d = hf * 8 + dd;                          // dims 0..15 (cols 0..143)
    float x = hat[p * 72 + 38 + d], q = qv[p * 80 + 38 + d];
    float mu[6], sg[6], pw[6];
    for (int k = 0; k < 3; ++k) {
      int b = p * 160 + d * 9 + k * 3;
      mu[k] = scr[b]; sg[k] = scr[b + 1]; pw[k] = scr[b + 2];
    }
    bitsO += gmm_bits(x, q, mu, sg, pw, 3);
  }
  __syncthreads();
  run_layer(hbuf, 128, 4, wt + OFF_WO2, 128, Ag.bo2, 270, 9, 8, lane, nullptr, 0, scr, 160);
  __syncthreads();
  for (int dd = 0; dd < 7; ++dd) {
    int d = 16 + hf * 7 + dd;                     // dims 16..29 (cols 144..269)
    float x = hat[p * 72 + 38 + d], q = qv[p * 80 + 38 + d];
    float mu[6], sg[6], pw[6];
    for (int k = 0; k < 3; ++k) {
      int b = p * 160 + (d * 9 - 144) + k * 3;
      mu[k] = scr[b]; sg[k] = scr[b + 1]; pw[k] = scr[b + 2];
    }
    bitsO += gmm_bits(x, q, mu, sg, pw, 3);
  }

  // ---- cross-lane reduce (2 lanes per point, wave32) and store ----
  bitsF += __shfl_xor(bitsF, 16, 32);
  bitsS += __shfl_xor(bitsS, 16, 32);
  bitsO += __shfl_xor(bitsO, 16, 32);
  int gp = pbase + p;
  if (lane < 16 && gp < n) {
    out[gp] = bitsF;
    out[(size_t)n + gp] = bitsS;
    out[(size_t)2 * n + gp] = bitsO;
  }
}

// ---------------- host entry ----------------
extern "C" void kernel_launch(void* const* d_in, const int* in_sizes, int n_in,
                              void* d_out, int out_size, void* d_ws, size_t ws_size,
                              hipStream_t stream) {
  const int n = in_sizes[0] / 48;   // hash_feats: N x 48
  auto F = [&](int i) { return (const float*)d_in[i]; };
  _Float16* wt = (_Float16*)d_ws;   // needs WT_TOTAL*2 = ~412KB of d_ws

  // d_in order: 0 hash,1 anchor,2 scaling,3 offset,4 u_feat,5 u_scaling,6 u_offset,
  // then params: 7 Wq1,8 bq1,9 Wq2,10 bq2,11 Wc1,12 bc1,13 Wc2,14 bc2,
  // 15 Ws1,16 bs1,17 Ws2,18 bs2,19 Wo1,20 bo1,21 Wo2,22 bo2,
  // then per i=1..3: Wa1_i,ba1_i,Wa2_i,ba2_i (23..34)
  PrepTable t;
  int ei = 0;
  auto add = [&](const float* src, int K, int N, int Kpad, size_t off) {
    int Npad = (N + 15) & ~15;
    t.e[ei++] = {src, K, N, Kpad, off, (size_t)Npad * (size_t)Kpad};
  };
  add(F(7), 48, 128, 64, OFF_WQ1);
  add(F(9), 128, 68, 128, OFF_WQ2);
  for (int c = 0; c < 4; ++c) add(F(11) + (size_t)c * 48 * 128, 48, 128, 64, OFF_WC1 + (size_t)c * 128 * 64);
  for (int c = 0; c < 4; ++c) add(F(13) + (size_t)c * 128 * 72, 128, 72, 128, OFF_WC2 + (size_t)c * 80 * 128);
  add(F(15), 48, 64, 64, OFF_WS1);
  add(F(17), 64, 54, 64, OFF_WS2);
  add(F(19), 48, 128, 64, OFF_WO1);
  add(F(21), 128, 270, 128, OFF_WO2);
  for (int i = 0; i < 3; ++i) add(F(23 + 4 * i), 48 + 8 * (i + 1), 128, 96, OFF_WA1 + (size_t)i * 128 * 96);
  for (int i = 0; i < 3; ++i) add(F(25 + 4 * i), 128, 72, 128, OFF_WA2 + (size_t)i * 80 * 128);

  int total = (int)WT_TOTAL;
  hac_prep_weights<<<(total + 255) / 256, 256, 0, stream>>>(t, wt, total);

  HacArgs a;
  a.hash = F(0); a.anchor = F(1); a.scaling = F(2); a.offset = F(3);
  a.u_feat = F(4); a.u_scaling = F(5); a.u_offset = F(6);
  a.bq1 = F(8); a.bq2 = F(10); a.bc1 = F(12); a.bc2 = F(14);
  a.bs1 = F(16); a.bs2 = F(18); a.bo1 = F(20); a.bo2 = F(22);
  for (int i = 0; i < 3; ++i) { a.ba1[i] = F(24 + 4 * i); a.ba2[i] = F(26 + 4 * i); }

  int grid = (n + 31) / 32;         // 32 points per block (2 waves x 16)
  size_t shbytes = 2 * (size_t)WAVE_BYTES;
  hac_main<<<grid, 64, shbytes, stream>>>(a, wt, (float*)d_out, n);
}
